// SNN_79611513799359
// MI455X (gfx1250) — compile-verified, matrix operand order checked by reading, not run on hardware
//
#include <hip/hip_runtime.h>

// ---------------------------------------------------------------------------
// SNN forward, batched over time:
//   K1: cur1 = X(32768x1024) @ W1^T (1024x2048) + b1 + bias1   [WMMA f16,
//       double-buffered LDS pipeline] -> staged into spk1 region of d_out
//   K2: LIF recurrence over T per (b,h) element, overwrites cur1 with spk1
//   K3: cur2 = spk1(32768x2048) @ W2^T (2048x10 pad 16) + b2   [WMMA]
//       -> staged into spk2 region of d_out
//   K4: LIF recurrence over T per (b,o) element, overwrites cur2 with spk2
// ---------------------------------------------------------------------------

typedef _Float16 half16  __attribute__((ext_vector_type(16)));
typedef _Float16 half8   __attribute__((ext_vector_type(8)));
typedef _Float16 half4   __attribute__((ext_vector_type(4)));
typedef float    float8  __attribute__((ext_vector_type(8)));
typedef float    float4v __attribute__((ext_vector_type(4)));

#define T_STEPS 128
#define BATCH   256
#define IN_DIM  1024
#define HID_DIM 2048
#define OUT_DIM 10
#define M1      (T_STEPS * BATCH)     // 32768 flattened (t,b) rows

#define LDS_STRIDE 40                 // halves per LDS row: 32 data + 8 pad

__device__ __forceinline__ float8 zero8() {
    float8 z = {0.f,0.f,0.f,0.f,0.f,0.f,0.f,0.f};
    return z;
}

// Build a 16-half A/B WMMA fragment from an LDS row (row-major K).
// lane L: halves [0..7] = k[hi8 .. hi8+7], halves [8..15] = k[16+hi8 ..]
__device__ __forceinline__ half16 frag_from_lds(const _Float16* rowBase, int hi8) {
    half8 lo = *(const half8*)(rowBase + hi8);
    half8 hi = *(const half8*)(rowBase + 16 + hi8);
    return __builtin_shufflevector(lo, hi, 0,1,2,3,4,5,6,7,8,9,10,11,12,13,14,15);
}

// Issue 4 x b128 global loads (16 contiguous f32) into registers.
__device__ __forceinline__ void gload16(const float* __restrict__ src, float4v r[4]) {
    r[0] = *(const float4v*)(src + 0);
    r[1] = *(const float4v*)(src + 4);
    r[2] = *(const float4v*)(src + 8);
    r[3] = *(const float4v*)(src + 12);
}

// Convert 16 staged f32 to f16 and store 32B to LDS.
__device__ __forceinline__ void cvtstore16(const float4v r[4], _Float16* dst) {
    half8 h0 = { (_Float16)r[0][0], (_Float16)r[0][1], (_Float16)r[0][2], (_Float16)r[0][3],
                 (_Float16)r[1][0], (_Float16)r[1][1], (_Float16)r[1][2], (_Float16)r[1][3] };
    half8 h1 = { (_Float16)r[2][0], (_Float16)r[2][1], (_Float16)r[2][2], (_Float16)r[2][3],
                 (_Float16)r[3][0], (_Float16)r[3][1], (_Float16)r[3][2], (_Float16)r[3][3] };
    *(half8*)(dst + 0) = h0;
    *(half8*)(dst + 8) = h1;
}

// ---------------- Kernel 1: big layer-1 GEMM (double-buffered) ----------------
// grid = (n-tiles=16, m-tiles=256): consecutive blocks share one A tile in L2.
__global__ __launch_bounds__(256) void snn_gemm1(
    const float* __restrict__ x, const float* __restrict__ W1,
    const float* __restrict__ b1, const float* __restrict__ bias1p,
    float* __restrict__ out1)
{
    __shared__ _Float16 As[2][128 * LDS_STRIDE];
    __shared__ _Float16 Bs[2][128 * LDS_STRIDE];

    const int tid  = threadIdx.x;
    const int wid  = tid >> 5;
    const int lane = tid & 31;
    const int r    = lane & 15;
    const int hi8  = (lane >> 4) * 8;

    const int nTile = blockIdx.x * 128;   // 16 n-tiles innermost -> A reuse in L2
    const int mTile = blockIdx.y * 128;
    const int mb = (wid & 1) * 64;        // wave M base within tile
    const int nb = (wid >> 1) * 32;       // wave N base within tile

    float8 acc[4][2];
    #pragma unroll
    for (int i = 0; i < 4; ++i)
        #pragma unroll
        for (int j = 0; j < 2; ++j) acc[i][j] = zero8();

    const int srow = tid >> 1;            // 128 staged rows, 2 threads/row
    const int ssel = (tid & 1) * 16;      // this thread stages 16 of 32 k-vals
    const float* aSrc = x  + (long)(mTile + srow) * IN_DIM + ssel;
    const float* bSrc = W1 + (long)(nTile + srow) * IN_DIM + ssel;
    _Float16* aDst0 = &As[0][srow * LDS_STRIDE + ssel];
    _Float16* bDst0 = &Bs[0][srow * LDS_STRIDE + ssel];
    _Float16* aDst1 = &As[1][srow * LDS_STRIDE + ssel];
    _Float16* bDst1 = &Bs[1][srow * LDS_STRIDE + ssel];

    // Prologue: stage tile k=0 into buffer 0.
    float4v ar[4], br[4];
    gload16(aSrc, ar);
    gload16(bSrc, br);
    cvtstore16(ar, aDst0);
    cvtstore16(br, bDst0);
    __syncthreads();

    int buf = 0;
    for (int kb = 32; kb <= IN_DIM; kb += 32) {
        const bool more = (kb < IN_DIM);
        if (more) {                        // prefetch next tile into registers
            gload16(aSrc + kb, ar);
            gload16(bSrc + kb, br);
        }

        // Compute on current buffer (covers the global-load latency).
        half16 af[4], bf[2];
        #pragma unroll
        for (int i = 0; i < 4; ++i)
            af[i] = frag_from_lds(&As[buf][(mb + i * 16 + r) * LDS_STRIDE], hi8);
        #pragma unroll
        for (int j = 0; j < 2; ++j)
            bf[j] = frag_from_lds(&Bs[buf][(nb + j * 16 + r) * LDS_STRIDE], hi8);
        #pragma unroll
        for (int i = 0; i < 4; ++i)
            #pragma unroll
            for (int j = 0; j < 2; ++j)
                acc[i][j] = __builtin_amdgcn_wmma_f32_16x16x32_f16(
                    false, af[i], false, bf[j], (short)0, acc[i][j], false, false);

        if (more) {                        // fill the other buffer, one barrier
            cvtstore16(ar, buf ? aDst0 : aDst1);
            cvtstore16(br, buf ? bDst0 : bDst1);
            __syncthreads();
            buf ^= 1;
        }
    }

    const float biasc = bias1p[0];
    #pragma unroll
    for (int i = 0; i < 4; ++i) {
        #pragma unroll
        for (int j = 0; j < 2; ++j) {
            const int gn   = nTile + nb + j * 16 + r;
            const float bb = b1[gn] + biasc;
            const int gm0  = mTile + mb + i * 16 + (lane >> 4) * 8;
            #pragma unroll
            for (int v = 0; v < 8; ++v)
                out1[(long)(gm0 + v) * HID_DIM + gn] = acc[i][j][v] + bb;
        }
    }
}

// ---------------- Kernel 2: layer-1 LIF recurrence (in place) ----------------
__global__ __launch_bounds__(256) void snn_lif1(float* __restrict__ out1)
{
    const long idx    = (long)blockIdx.x * 256 + threadIdx.x;  // < B*H
    const long stride = (long)BATCH * HID_DIM;
    float mem = 0.0f;
    for (int t = 0; t < T_STEPS; ++t) {
        const float cur   = out1[(long)t * stride + idx];
        const float reset = (mem > 1.0f) ? 1.0f : 0.0f;    // uses previous mem
        mem = 0.9f * mem + cur - reset;                    // BETA1, THR=1
        out1[(long)t * stride + idx] = (mem > 1.0f) ? 1.0f : 0.0f;  // spike
    }
}

// ---------------- Kernel 3: layer-2 GEMM (N padded 10->16) ----------------
__global__ __launch_bounds__(256) void snn_gemm2(
    const float* __restrict__ spk1, const float* __restrict__ W2,
    const float* __restrict__ b2, float* __restrict__ out2)
{
    __shared__ _Float16 As[2][128 * LDS_STRIDE];
    __shared__ _Float16 Ws[2][16 * LDS_STRIDE];

    const int tid  = threadIdx.x;
    const int wid  = tid >> 5;
    const int lane = tid & 31;
    const int r    = lane & 15;
    const int hi8  = (lane >> 4) * 8;

    const int mTile = blockIdx.x * 128;
    const int mb    = wid * 16;           // each of 8 waves owns 16 M rows

    float8 acc = zero8();
    const int srow = tid >> 1;
    const int ssel = (tid & 1) * 16;
    const float* aSrc = spk1 + (long)(mTile + srow) * HID_DIM + ssel;

    const int wrow = tid >> 3;            // 0..15 (for tid<128)
    const int wkk  = (tid & 7) * 4;       // 0..28
    const float* wSrc = W2 + (long)(wrow < OUT_DIM ? wrow : 0) * HID_DIM + wkk;
    const bool wvalid = (tid < 128) && (wrow < OUT_DIM);

    // Prologue: stage tile k=0 into buffer 0.
    float4v ar[4], wr = {0.f,0.f,0.f,0.f};
    gload16(aSrc, ar);
    if (wvalid) wr = *(const float4v*)(wSrc);
    cvtstore16(ar, &As[0][srow * LDS_STRIDE + ssel]);
    if (tid < 128) {
        half4 h = { (_Float16)wr[0], (_Float16)wr[1], (_Float16)wr[2], (_Float16)wr[3] };
        *(half4*)(&Ws[0][wrow * LDS_STRIDE + wkk]) = h;
    }
    __syncthreads();

    int buf = 0;
    for (int kb = 32; kb <= HID_DIM; kb += 32) {
        const bool more = (kb < HID_DIM);
        if (more) {
            gload16(aSrc + kb, ar);
            if (wvalid) wr = *(const float4v*)(wSrc + kb);
        }

        half16 af = frag_from_lds(&As[buf][(mb + r) * LDS_STRIDE], hi8);
        half16 bf = frag_from_lds(&Ws[buf][r * LDS_STRIDE], hi8);
        acc = __builtin_amdgcn_wmma_f32_16x16x32_f16(
            false, af, false, bf, (short)0, acc, false, false);

        if (more) {
            cvtstore16(ar, &As[buf ^ 1][srow * LDS_STRIDE + ssel]);
            if (tid < 128) {
                half4 h = { (_Float16)wr[0], (_Float16)wr[1],
                            (_Float16)wr[2], (_Float16)wr[3] };
                *(half4*)(&Ws[buf ^ 1][wrow * LDS_STRIDE + wkk]) = h;
            }
            __syncthreads();
            buf ^= 1;
        }
    }

    if (r < OUT_DIM) {
        const float bb = b2[r];
        const int gm0  = mTile + mb + (lane >> 4) * 8;
        #pragma unroll
        for (int v = 0; v < 8; ++v)
            out2[(long)(gm0 + v) * OUT_DIM + r] = acc[v] + bb;
    }
}

// ---------------- Kernel 4: layer-2 LIF recurrence (in place) ----------------
__global__ __launch_bounds__(256) void snn_lif2(float* __restrict__ out2)
{
    const int idx = blockIdx.x * 256 + threadIdx.x;
    if (idx >= BATCH * OUT_DIM) return;
    const int stride = BATCH * OUT_DIM;
    float mem = 0.0f;
    for (int t = 0; t < T_STEPS; ++t) {
        const float cur   = out2[t * stride + idx];
        const float reset = (mem > 1.0f) ? 1.0f : 0.0f;
        mem = 0.8f * mem + cur - reset;                    // BETA2
        out2[t * stride + idx] = (mem > 1.0f) ? 1.0f : 0.0f;
    }
}

// ---------------------------------------------------------------------------
extern "C" void kernel_launch(void* const* d_in, const int* in_sizes, int n_in,
                              void* d_out, int out_size, void* d_ws, size_t ws_size,
                              hipStream_t stream) {
    (void)in_sizes; (void)n_in; (void)out_size; (void)d_ws; (void)ws_size;
    const float* x     = (const float*)d_in[0];  // (128,256,1024)
    const float* W1    = (const float*)d_in[1];  // (2048,1024)
    const float* b1    = (const float*)d_in[2];  // (2048,)
    const float* bias1 = (const float*)d_in[3];  // scalar
    const float* W2    = (const float*)d_in[4];  // (10,2048)
    const float* b2    = (const float*)d_in[5];  // (10,)

    float* out1 = (float*)d_out;                               // spk1 (T,B,H)
    float* out2 = out1 + (long)T_STEPS * BATCH * HID_DIM;      // spk2 (T,B,O)

    dim3 gA(HID_DIM / 128, M1 / 128);   // (16 n-tiles, 256 m-tiles)
    snn_gemm1<<<gA, 256, 0, stream>>>(x, W1, b1, bias1, out1);
    snn_lif1<<<(BATCH * HID_DIM) / 256, 256, 0, stream>>>(out1);
    snn_gemm2<<<M1 / 128, 256, 0, stream>>>(out1, W2, b2, out2);
    snn_lif2<<<(BATCH * OUT_DIM + 255) / 256, 256, 0, stream>>>(out2);
}
